// SingleHead_76295799046550
// MI455X (gfx1250) — matrix operand order repaired and checked by reading, not compiled, not run on hardware
//
#include <hip/hip_runtime.h>

// ---------------- problem constants ----------------
#define CCH     128      // channels (K and N of the big GEMMs)
#define TILE_M  64       // rows per block tile
#define LDK     136      // padded LDS k-stride (f16 elems); 136*2=272B = 17*16B
#define NB_GEMM 1024     // persistent GEMM blocks
#define NB_POOL 512      // pooling blocks
#define NSEG    16
#define OUTC    64
#define BN_EPS  1e-5f

typedef __attribute__((ext_vector_type(16))) _Float16 v16h;
typedef __attribute__((ext_vector_type(8)))  _Float16 v8h;
typedef __attribute__((ext_vector_type(8)))  float    v8f;

__device__ __forceinline__ v16h concat8(v8h lo, v8h hi) {
  return __builtin_shufflevector(lo, hi, 0,1,2,3,4,5,6,7,8,9,10,11,12,13,14,15);
}

// A-matrix 16x32 f16 fragment (lanes0-15 hold K 0..15,
// lanes16-31 hold K 16..31); K-contiguous LDS -> two b128 loads.
__device__ __forceinline__ v16h ldsFragA(const _Float16* row, int kbase, int hf) {
  v8h lo = *(const v8h*)(row + kbase + 16 * hf);
  v8h hi = *(const v8h*)(row + kbase + 16 * hf + 8);
  return concat8(lo, hi);
}
// B-matrix 32x16 f16 fragment (lanes0-15: K=0..15, lanes16-31: K=16..31,
// col N = lane%16); W stored transposed in LDS (Wt[n][k]) -> two b128 loads.
__device__ __forceinline__ v16h ldsFragB(const _Float16* col, int kbase, int hf) {
  v8h lo = *(const v8h*)(col + kbase + 16 * hf);
  v8h hi = *(const v8h*)(col + kbase + 16 * hf + 8);
  return concat8(lo, hi);
}

// ------------------------------------------------------------------
// Fused GEMM: H(f16) = act(A) @ W + bias, plus per-channel sum/sumsq
// partials (f32, from the f32 WMMA accumulators) for BatchNorm.
// act = identity reading f32 (layer0) or relu(a*scale+shift) reading
// f16 (layer1, BN0+ReLU folded into the A-tile load).
// ------------------------------------------------------------------
__global__ __launch_bounds__(256) void gemm_bn_kernel(
    const float* __restrict__ Af32, const _Float16* __restrict__ Ah16,
    const float* __restrict__ Wg, const float* __restrict__ biasg,
    const float* __restrict__ scaleIn, const float* __restrict__ shiftIn,
    int applyBN, _Float16* __restrict__ Hout,
    float* __restrict__ statsPartial, int nRows)
{
  __shared__ __align__(16) _Float16 Alds[TILE_M * LDK];
  __shared__ __align__(16) _Float16 Wlds[CCH * LDK];   // Wlds[n*LDK+k] = W[k][n]
  __shared__ float biasLds[CCH], scaleLds[CCH], shiftLds[CCH];
  __shared__ float sumLds[CCH], sqLds[CCH];

  const int t = threadIdx.x;

  for (int i = t; i < CCH * CCH; i += 256) {
    int k = i / CCH, n = i % CCH;
    Wlds[n * LDK + k] = (_Float16)Wg[i];
  }
  if (t < CCH) {
    biasLds[t]  = biasg[t];
    scaleLds[t] = applyBN ? scaleIn[t] : 1.0f;
    shiftLds[t] = applyBN ? shiftIn[t] : 0.0f;
    sumLds[t] = 0.0f; sqLds[t] = 0.0f;
  }
  __syncthreads();

  const int lane   = t & 31;
  const int hf     = lane >> 4;      // wave half
  const int l16    = lane & 15;
  const int wv     = t >> 5;         // 8 waves
  const int strip  = wv & 3;         // 16-row strip within tile
  const int ctBase = (wv >> 2) * 4;  // 4 col-tiles of 16 per wave

  // W is loop-invariant: hoist all 16 B fragments into registers.
  v16h Bf[4][4];
#pragma unroll
  for (int cti = 0; cti < 4; ++cti) {
    const _Float16* Bcol = Wlds + ((ctBase + cti) * 16 + l16) * LDK;
#pragma unroll
    for (int ks = 0; ks < 4; ++ks) Bf[cti][ks] = ldsFragB(Bcol, ks * 32, hf);
  }

  const int numTiles = (nRows + TILE_M - 1) / TILE_M;
  for (int tile = blockIdx.x; tile < numTiles; tile += gridDim.x) {
    const int rowBase = tile * TILE_M;
    const bool fullTile = (rowBase + TILE_M) <= nRows;

    // stage A tile -> f16 LDS (BN0+ReLU folded for layer1)
    if (!applyBN) {
      for (int i = t; i < TILE_M * (CCH / 4); i += 256) {
        int r  = i / (CCH / 4);
        int c4 = (i % (CCH / 4)) * 4;
        int gr = rowBase + r;
        float4 v = make_float4(0.f, 0.f, 0.f, 0.f);
        if (gr < nRows) v = *(const float4*)(Af32 + (size_t)gr * CCH + c4);
        _Float16* dst = Alds + r * LDK + c4;
        dst[0] = (_Float16)v.x; dst[1] = (_Float16)v.y;
        dst[2] = (_Float16)v.z; dst[3] = (_Float16)v.w;
      }
    } else {
      for (int i = t; i < TILE_M * (CCH / 8); i += 256) {
        int r  = i / (CCH / 8);
        int c8 = (i % (CCH / 8)) * 8;
        int gr = rowBase + r;
        v8h v = {};
        if (gr < nRows) v = *(const v8h*)(Ah16 + (size_t)gr * CCH + c8);
        _Float16* dst = Alds + r * LDK + c8;
#pragma unroll
        for (int e = 0; e < 8; ++e) {
          float x = fmaxf((float)v[e] * scaleLds[c8 + e] + shiftLds[c8 + e], 0.f);
          dst[e] = (_Float16)x;
        }
      }
    }
    __syncthreads();

    const _Float16* Arow = Alds + (strip * 16 + l16) * LDK;
    v16h a0 = ldsFragA(Arow,  0, hf);
    v16h a1 = ldsFragA(Arow, 32, hf);
    v16h a2 = ldsFragA(Arow, 64, hf);
    v16h a3 = ldsFragA(Arow, 96, hf);

#pragma unroll
    for (int cti = 0; cti < 4; ++cti) {
      const int n = (ctBase + cti) * 16 + l16;
      v8f acc = {};
      acc = __builtin_amdgcn_wmma_f32_16x16x32_f16(false, a0, false, Bf[cti][0], (short)0, acc, false, false);
      acc = __builtin_amdgcn_wmma_f32_16x16x32_f16(false, a1, false, Bf[cti][1], (short)0, acc, false, false);
      acc = __builtin_amdgcn_wmma_f32_16x16x32_f16(false, a2, false, Bf[cti][2], (short)0, acc, false, false);
      acc = __builtin_amdgcn_wmma_f32_16x16x32_f16(false, a3, false, Bf[cti][3], (short)0, acc, false, false);

      const float bn = biasLds[n];
      float s = 0.f, q = 0.f;
      const int rB = rowBase + strip * 16 + hf * 8;  // C/D: vgpr j -> row j + 8*half
      if (fullTile) {                                // branchless fast path
        _Float16* hp = Hout + (size_t)rB * CCH + n;
#pragma unroll
        for (int j = 0; j < 8; ++j) {
          float hval = acc[j] + bn;
          hp[(size_t)j * CCH] = (_Float16)hval;
          s += hval; q += hval * hval;
        }
      } else {                                       // only the very last tile
#pragma unroll
        for (int j = 0; j < 8; ++j) {
          int gr = rB + j;
          if (gr < nRows) {
            float hval = acc[j] + bn;
            Hout[(size_t)gr * CCH + n] = (_Float16)hval;
            s += hval; q += hval * hval;
          }
        }
      }
      atomicAdd(&sumLds[n], s);   // ds_add_f32
      atomicAdd(&sqLds[n], q);
    }
    __syncthreads();
  }

  if (t < CCH) {
    statsPartial[(size_t)blockIdx.x * (2 * CCH) + t]       = sumLds[t];
    statsPartial[(size_t)blockIdx.x * (2 * CCH) + CCH + t] = sqLds[t];
  }
}

// Deterministic fixed-order BN stats reduction -> folded scale/shift.
__global__ void stats_finalize_kernel(const float* __restrict__ statsPartial, int nb,
                                      const float* __restrict__ gamma,
                                      const float* __restrict__ beta,
                                      float* __restrict__ scaleOut,
                                      float* __restrict__ shiftOut, float invN)
{
  int c = threadIdx.x;  // 128 threads
  float s = 0.f, q = 0.f;
  for (int b = 0; b < nb; ++b) {
    s += statsPartial[(size_t)b * (2 * CCH) + c];
    q += statsPartial[(size_t)b * (2 * CCH) + CCH + c];
  }
  float m  = s * invN;
  float v  = fmaxf(q * invN - m * m, 0.f);     // biased var (torch BN)
  float rs = rsqrtf(v + BN_EPS);
  float sc = rs * gamma[c];
  scaleOut[c] = sc;
  shiftOut[c] = beta[c] - m * sc;
}

__global__ void zero_kernel(float* p, int n) {
  int i = blockIdx.x * blockDim.x + threadIdx.x;
  if (i < n) p[i] = 0.f;
}

// Segment max/sum pooling with BN1+ReLU folded in. relu(x) >= 0 so
// float-bit integer atomicMax is exact and order-independent.
__global__ __launch_bounds__(256) void pool_kernel(
    const _Float16* __restrict__ H, const float* __restrict__ scale,
    const float* __restrict__ shift, const int* __restrict__ offset,
    float* __restrict__ segMax, float* __restrict__ segSumPartial, int nRows)
{
  __shared__ int   maxLds[NSEG * CCH];
  __shared__ float sumLds[NSEG * CCH];
  __shared__ int   offs[NSEG];
  const int t = threadIdx.x;
  if (t < NSEG) offs[t] = offset[t];
  for (int i = t; i < NSEG * CCH; i += 256) { maxLds[i] = 0; sumLds[i] = 0.f; }
  __syncthreads();

  const int c  = t & (CCH - 1);
  const int rp = t >> 7;                       // two rows in flight per block
  const int rowsPer = (nRows + gridDim.x - 1) / gridDim.x;
  const int r0 = blockIdx.x * rowsPer;
  const int r1 = min(r0 + rowsPer, nRows);
  const float sc = scale[c], sh = shift[c];

  int curSeg = 0;
  int segEnd = offs[0];
  float runMax = 0.f, runSum = 0.f;
  for (int r = r0 + rp; r < r1; r += 2) {
    while (r >= segEnd) {                      // offs[NSEG-1]==nRows bounds this
      atomicMax(&maxLds[curSeg * CCH + c], __float_as_int(runMax));
      atomicAdd(&sumLds[curSeg * CCH + c], runSum);
      runMax = 0.f; runSum = 0.f; ++curSeg;
      segEnd = offs[curSeg];
    }
    float x = fmaxf((float)H[(size_t)r * CCH + c] * sc + sh, 0.f);
    runMax = fmaxf(runMax, x);
    runSum += x;
  }
  if (curSeg < NSEG) {
    atomicMax(&maxLds[curSeg * CCH + c], __float_as_int(runMax));
    atomicAdd(&sumLds[curSeg * CCH + c], runSum);
  }
  __syncthreads();

  for (int i = t; i < NSEG * CCH; i += 256) {
    atomicMax((int*)segMax + i, maxLds[i]);
    segSumPartial[(size_t)blockIdx.x * (NSEG * CCH) + i] = sumLds[i];
  }
}

__global__ void reduce_sum_kernel(const float* __restrict__ partial, int nb,
                                  float* __restrict__ outp)
{
  int i = blockIdx.x * blockDim.x + threadIdx.x;
  if (i < NSEG * CCH) {
    float s = 0.f;
    for (int b = 0; b < nb; ++b) s += partial[(size_t)b * (NSEG * CCH) + i];
    outp[i] = s;
  }
}

// pooled = [segMax | segSum/count]  (16 x 256); out = pooled @ W_out + b_out
__global__ __launch_bounds__(256) void final_kernel(
    const float* __restrict__ segMaxF, const float* __restrict__ segSum,
    const int* __restrict__ offset, const float* __restrict__ Wout,
    const float* __restrict__ bout, float* __restrict__ out)
{
  __shared__ float pooled[NSEG * 2 * CCH];
  __shared__ float cnt[NSEG];
  const int t = threadIdx.x;
  if (t < NSEG) {
    int e = offset[t];
    int s = (t == 0) ? 0 : offset[t - 1];
    cnt[t] = (float)max(e - s, 1);
  }
  __syncthreads();
  for (int i = t; i < NSEG * CCH; i += 256) {
    int s = i >> 7, c = i & (CCH - 1);
    pooled[s * 2 * CCH + c]       = segMaxF[i];          // empty seg -> 0 by init
    pooled[s * 2 * CCH + CCH + c] = segSum[i] / cnt[s];
  }
  __syncthreads();
  for (int i = t; i < NSEG * OUTC; i += 256) {
    int s = i >> 6, o = i & (OUTC - 1);
    float acc = bout[o];
    for (int j = 0; j < 2 * CCH; ++j) acc += pooled[s * 2 * CCH + j] * Wout[j * OUTC + o];
    out[i] = acc;
  }
}

// ------------------------------------------------------------------
extern "C" void kernel_launch(void* const* d_in, const int* in_sizes, int n_in,
                              void* d_out, int out_size, void* d_ws, size_t ws_size,
                              hipStream_t stream)
{
  const float* feat   = (const float*)d_in[0];
  const int*   offset = (const int*)  d_in[1];   // JAX default x64-off -> int32
  const float* W      = (const float*)d_in[2];   // [2,128,128]
  const float* bias   = (const float*)d_in[3];   // [2,128]
  const float* gamma  = (const float*)d_in[4];
  const float* beta   = (const float*)d_in[5];
  const float* Wout   = (const float*)d_in[6];   // [256,64]
  const float* bout   = (const float*)d_in[7];   // [64]
  float* out = (float*)d_out;

  const int N = in_sizes[0] / CCH;

  char* ws = (char*)d_ws;
  _Float16* h = (_Float16*)ws;                         // N*128 f16 (in-place layer2)
  size_t off = (size_t)N * CCH * sizeof(_Float16);
  off = (off + 255) & ~(size_t)255;
  float* statsPartial = (float*)(ws + off); off += (size_t)NB_GEMM * 2 * CCH * sizeof(float);
  float* scale0 = (float*)(ws + off); off += CCH * sizeof(float);
  float* shift0 = (float*)(ws + off); off += CCH * sizeof(float);
  float* scale1 = (float*)(ws + off); off += CCH * sizeof(float);
  float* shift1 = (float*)(ws + off); off += CCH * sizeof(float);
  float* segMax = (float*)(ws + off); off += NSEG * CCH * sizeof(float);
  float* segSum = (float*)(ws + off); off += NSEG * CCH * sizeof(float);
  float* segSumPartial = (float*)(ws + off);           // NB_POOL*2048 f32

  const float invN = 1.0f / (float)N;

  zero_kernel<<<(NSEG * CCH + 255) / 256, 256, 0, stream>>>(segMax, NSEG * CCH);

  gemm_bn_kernel<<<NB_GEMM, 256, 0, stream>>>(feat, nullptr, W, bias,
                                              nullptr, nullptr, 0,
                                              h, statsPartial, N);
  stats_finalize_kernel<<<1, CCH, 0, stream>>>(statsPartial, NB_GEMM, gamma, beta,
                                               scale0, shift0, invN);
  gemm_bn_kernel<<<NB_GEMM, 256, 0, stream>>>(nullptr, h, W + CCH * CCH, bias + CCH,
                                              scale0, shift0, 1,
                                              h, statsPartial, N);
  stats_finalize_kernel<<<1, CCH, 0, stream>>>(statsPartial, NB_GEMM, gamma + CCH,
                                               beta + CCH, scale1, shift1, invN);
  pool_kernel<<<NB_POOL, 256, 0, stream>>>(h, scale1, shift1, offset, segMax,
                                           segSumPartial, N);
  reduce_sum_kernel<<<(NSEG * CCH + 255) / 256, 256, 0, stream>>>(segSumPartial,
                                                                  NB_POOL, segSum);
  final_kernel<<<1, 256, 0, stream>>>(segMax, segSum, offset, Wout, bout, out);
}
